// XlModel_85495618995076
// MI455X (gfx1250) — matrix operand order repaired and checked by reading, not compiled, not run on hardware
//
#include <hip/hip_runtime.h>
#include <hip/hip_bf16.h>
#include <math.h>

// ---------------- types ----------------
typedef __attribute__((ext_vector_type(16))) _Float16 v16h;
typedef __attribute__((ext_vector_type(8)))  _Float16 v8h;
typedef __attribute__((ext_vector_type(4)))  _Float16 v4h;
typedef __attribute__((ext_vector_type(8)))  float    v8f;

// ---------------- model constants ----------------
#define MB   16      // batch
#define MS   512     // sequence
#define MF_  256     // input features
#define MD   512     // model dim
#define MH   8       // heads
#define MDH  64      // head dim
#define ML   4       // layers
#define MMEM 256
#define MNI  32
#define MNCS 64
#define MOUT 1000

// GEMM tile config
#define BM 128
#define BN 128
#define BK 32

enum {
  FRELU = 1, FBIAS = 4, FRES = 8,
  FA16 = 16,    // A operand is f16
  FB16 = 32,    // B operand is f16
  FC16 = 64,    // store C as f16
  FTRANSC = 128 // store C transposed: C[col*ldc + row]
};

struct GemmP {
  const void* A; const void* Bm; const float* bias; const float* res; void* C;
  int M, N, K;
  int lda, ldb, ldc, ldres;
  long long sAo, sAi, sBo, sBi, sCo, sCi, sRo, sRi;
  int inner;     // inner batch count (heads); total batches = gridDim.z
  float alpha;
  int flags;
};

// ============================================================
// Batched-strided GEMM: C = epilogue(alpha * A[M][K] @ B[N][K]^T)
// Both operands K-contiguous (weights pre-transposed to [N][K]).
// Dtypes are template params -> branch-free steady-state loop.
// f16/f32 -> f16 LDS tiles (double buffered, register-staged)
// -> v_wmma_f32_16x16x32_f16 -> f32 accum -> bias/relu/res epilogue.
// OOB rows are address-clamped (their C rows/cols are never stored).
// ============================================================
template <bool TA16, bool TB16>
__global__ __launch_bounds__(256) void k_gemm(GemmP p) {
  __shared__ _Float16 As[2][BM][BK];   // [m][k], stride 64B
  __shared__ _Float16 Bs[2][BN][BK];   // [n][k]

  const int bz    = blockIdx.z;
  const int outer = bz / p.inner;
  const int innr  = bz % p.inner;
  const long long offA = outer * p.sAo + innr * p.sAi;
  const long long offB = outer * p.sBo + innr * p.sBi;
  const long long offC = outer * p.sCo + innr * p.sCi;
  const float*    A32 = (const float*)p.A + offA;
  const _Float16* A16 = (const _Float16*)p.A + offA;
  const float*    B32 = (const float*)p.Bm + offB;
  const _Float16* B16 = (const _Float16*)p.Bm + offB;

  const int m0 = blockIdx.y * BM;
  const int n0 = blockIdx.x * BN;
  const int t    = threadIdx.x;
  const int lane = t & 31;
  const int wid  = t >> 5;
  const int wm   = (wid & 3) * 32;   // wave row base (4 wave-rows)
  const int wn   = (wid >> 2) * 64;  // wave col base (2 wave-cols)
  const int Mc = p.M - 1, Nc = p.N - 1;

  v8f acc[8] = {};

  // register staging for the next tile
  float4 aF[4], bF[4];
  v8h    aH[2], bH[2];

  auto fetchA = [&](int k0) {
    if constexpr (TA16) {
      #pragma unroll
      for (int j = 0; j < 2; j++) {
        int fi = t + j * 256;
        int r  = min(m0 + (fi >> 2), Mc);
        aH[j] = *(const v8h*)(A16 + (long long)r * p.lda + k0 + (fi & 3) * 8);
      }
    } else {
      #pragma unroll
      for (int j = 0; j < 4; j++) {
        int fi = t + j * 256;
        int r  = min(m0 + (fi >> 3), Mc);
        aF[j] = *(const float4*)(A32 + (long long)r * p.lda + k0 + (fi & 7) * 4);
      }
    }
  };
  auto fetchB = [&](int k0) {
    if constexpr (TB16) {
      #pragma unroll
      for (int j = 0; j < 2; j++) {
        int fi = t + j * 256;
        int r  = min(n0 + (fi >> 2), Nc);
        bH[j] = *(const v8h*)(B16 + (long long)r * p.ldb + k0 + (fi & 3) * 8);
      }
    } else {
      #pragma unroll
      for (int j = 0; j < 4; j++) {
        int fi = t + j * 256;
        int r  = min(n0 + (fi >> 3), Nc);
        bF[j] = *(const float4*)(B32 + (long long)r * p.ldb + k0 + (fi & 7) * 4);
      }
    }
  };
  auto commitA = [&](int bsel) {
    if constexpr (TA16) {
      #pragma unroll
      for (int j = 0; j < 2; j++) {
        int fi = t + j * 256;
        *(v8h*)&As[bsel][fi >> 2][(fi & 3) * 8] = aH[j];
      }
    } else {
      #pragma unroll
      for (int j = 0; j < 4; j++) {
        int fi = t + j * 256;
        v4h h; h[0] = (_Float16)aF[j].x; h[1] = (_Float16)aF[j].y;
               h[2] = (_Float16)aF[j].z; h[3] = (_Float16)aF[j].w;
        *(v4h*)&As[bsel][fi >> 3][(fi & 7) * 4] = h;
      }
    }
  };
  auto commitB = [&](int bsel) {
    if constexpr (TB16) {
      #pragma unroll
      for (int j = 0; j < 2; j++) {
        int fi = t + j * 256;
        *(v8h*)&Bs[bsel][fi >> 2][(fi & 3) * 8] = bH[j];
      }
    } else {
      #pragma unroll
      for (int j = 0; j < 4; j++) {
        int fi = t + j * 256;
        v4h h; h[0] = (_Float16)bF[j].x; h[1] = (_Float16)bF[j].y;
               h[2] = (_Float16)bF[j].z; h[3] = (_Float16)bF[j].w;
        *(v4h*)&Bs[bsel][fi >> 3][(fi & 7) * 4] = h;
      }
    }
  };

  const int ml = lane & 15;   // A: M row / B: N col within frag
  const int kh = lane >> 4;   // A: K-half selector / B: K-group selector

  auto compute = [&](int bsel) {
    // fragments (ISA 7.12.2 layouts) + 8 WMMAs on current buffer
    v16h afrag[2];
    #pragma unroll
    for (int mf = 0; mf < 2; mf++) {
      v8h* fp = (v8h*)&afrag[mf];
      const _Float16* base = &As[bsel][wm + mf * 16 + ml][kh * 8];
      fp[0] = *(const v8h*)base;          // K = kh*8 + 0..7
      fp[1] = *(const v8h*)(base + 16);   // K = 16 + kh*8 + 0..7
    }
    v16h bfrag[4];
    #pragma unroll
    for (int nf = 0; nf < 4; nf++) {
      v8h* fp = (v8h*)&bfrag[nf];
      const _Float16* base = &Bs[bsel][wn + nf * 16 + ml][kh * 8];
      fp[0] = *(const v8h*)base;          // K = kh*8 + 0..7
      fp[1] = *(const v8h*)(base + 16);   // K = 16 + kh*8 + 0..7
    }
    #pragma unroll
    for (int mf = 0; mf < 2; mf++)
      #pragma unroll
      for (int nf = 0; nf < 4; nf++)
        acc[mf * 4 + nf] = __builtin_amdgcn_wmma_f32_16x16x32_f16(
            false, afrag[mf], false, bfrag[nf], (short)0, acc[mf * 4 + nf],
            false, false);
  };

  // ---- software pipeline: fetch k+1 -> compute k -> commit k+1 -> barrier
  fetchA(0); fetchB(0);
  commitA(0); commitB(0);
  __syncthreads();

  int bsel = 0;
  int k0 = 0;
  for (; k0 + BK < p.K; k0 += BK) {
    fetchA(k0 + BK); fetchB(k0 + BK);   // global loads overlap compute
    compute(bsel);
    commitA(bsel ^ 1); commitB(bsel ^ 1);
    __syncthreads();
    bsel ^= 1;
  }
  compute(bsel);                         // tail iteration

  // ---- epilogue: C/D layout: lane = (m>=8)*16 + n, vgpr = m%8
  const int nl = lane & 15;
  const int mh = (lane >> 4) * 8;
  float*     C32 = (float*)p.C + offC;
  _Float16*  C16 = (_Float16*)p.C + offC;
  const float* R = (p.flags & FRES) ? (p.res + outer * p.sRo + innr * p.sRi) : nullptr;
  #pragma unroll
  for (int mf = 0; mf < 2; mf++)
    #pragma unroll
    for (int nf = 0; nf < 4; nf++) {
      v8f a = acc[mf * 4 + nf];
      #pragma unroll
      for (int r = 0; r < 8; r++) {
        int row = m0 + wm + mf * 16 + mh + r;
        int col = n0 + wn + nf * 16 + nl;
        if (row < p.M && col < p.N) {
          float v = p.alpha * a[r];
          if (p.flags & FBIAS) v += p.bias[col];
          if (p.flags & FRELU) v = fmaxf(v, 0.f);
          if (p.flags & FRES)  v += R[(long long)row * p.ldres + col];
          long long idx = (p.flags & FTRANSC) ? ((long long)col * p.ldc + row)
                                              : ((long long)row * p.ldc + col);
          if (p.flags & FC16) C16[idx] = (_Float16)v;
          else                C32[idx] = v;
        }
      }
    }
}

// ============================================================
// LayerNorm over rows of length D, f32 in -> f16 out (feeds GEMM A-side)
// ============================================================
__global__ __launch_bounds__(128) void k_layernorm16(const float* __restrict__ x,
                                                     const float* __restrict__ g,
                                                     const float* __restrict__ b,
                                                     _Float16* __restrict__ y,
                                                     int D) {
  long long row = blockIdx.x;
  const float* xr = x + row * D;
  _Float16*    yr = y + row * D;
  int t = threadIdx.x;
  float s = 0.f, s2 = 0.f;
  for (int i = t; i < D; i += 128) { float v = xr[i]; s += v; s2 += v * v; }
  __shared__ float sh[128], sh2[128];
  sh[t] = s; sh2[t] = s2; __syncthreads();
  for (int off = 64; off > 0; off >>= 1) {
    if (t < off) { sh[t] += sh[t + off]; sh2[t] += sh2[t + off]; }
    __syncthreads();
  }
  float mean = sh[0] / (float)D;
  float var  = sh2[0] / (float)D - mean * mean;
  float inv  = rsqrtf(var + 0.001f);
  for (int i = t; i < D; i += 128)
    yr[i] = (_Float16)(g[i] * (xr[i] - mean) * inv + b[i]);
}

// ============================================================
// Row softmax, f32 (in may equal out)
// ============================================================
__global__ __launch_bounds__(256) void k_softmax(const float* __restrict__ in,
                                                 float* __restrict__ out, int n) {
  long long row = blockIdx.x;
  const float* xr = in + row * n;
  float*       yr = out + row * n;
  int t = threadIdx.x;
  __shared__ float sh[256];
  float m = -3.4e38f;
  for (int i = t; i < n; i += 256) m = fmaxf(m, xr[i]);
  sh[t] = m; __syncthreads();
  for (int off = 128; off > 0; off >>= 1) {
    if (t < off) sh[t] = fmaxf(sh[t], sh[t + off]);
    __syncthreads();
  }
  m = sh[0]; __syncthreads();
  float s = 0.f;
  for (int i = t; i < n; i += 256) { float e = __expf(xr[i] - m); yr[i] = e; s += e; }
  sh[t] = s; __syncthreads();
  for (int off = 128; off > 0; off >>= 1) {
    if (t < off) sh[t] += sh[t + off];
    __syncthreads();
  }
  float inv = 1.f / sh[0];
  for (int i = t; i < n; i += 256) yr[i] *= inv;
}

// ============================================================
// Weight pre-transpose: WT[n*K + k] = (f16) W[k*N + n]
// ============================================================
__global__ __launch_bounds__(256) void k_transpose16(const float* __restrict__ W,
                                                     _Float16* __restrict__ WT,
                                                     int K, int N) {
  long long idx = (long long)blockIdx.x * 256 + threadIdx.x;
  if (idx >= (long long)K * N) return;
  int k = (int)(idx % K);
  int n = (int)(idx / K);
  WT[idx] = (_Float16)W[(long long)k * N + n];
}

// ============================================================
// Copy a (B, R, D) row block into a batched destination at a row offset
// ============================================================
__global__ void k_copy_rows(float* __restrict__ dst, long long dstBS, int dstRowOff,
                            const float* __restrict__ src, long long srcBS,
                            int Bn, int R, int D) {
  long long idx = (long long)blockIdx.x * blockDim.x + threadIdx.x;
  long long total = (long long)Bn * R * D;
  if (idx >= total) return;
  int d = (int)(idx % D);
  long long tmp = idx / D;
  int r = (int)(tmp % R);
  int b = (int)(tmp / R);
  dst[b * dstBS + (long long)(dstRowOff + r) * D + d] =
      src[b * srcBS + (long long)r * D + d];
}

// ============================================================
// Loss: sum of squared differences (atomic), then sqrt-store
// ============================================================
__global__ void k_zero1(float* p) { *p = 0.f; }

__global__ __launch_bounds__(256) void k_sumsq_diff(const float* __restrict__ a,
                                                    const float* __restrict__ b,
                                                    float* __restrict__ acc,
                                                    long long n) {
  long long i = (long long)blockIdx.x * blockDim.x + threadIdx.x;
  long long stride = (long long)gridDim.x * blockDim.x;
  float s = 0.f;
  for (; i < n; i += stride) { float d = a[i] - b[i]; s += d * d; }
  __shared__ float sh[256];
  sh[threadIdx.x] = s; __syncthreads();
  for (int off = 128; off > 0; off >>= 1) {
    if (threadIdx.x < off) sh[threadIdx.x] += sh[threadIdx.x + off];
    __syncthreads();
  }
  if (threadIdx.x == 0) atomicAdd(acc, sh[0]);
}

__global__ void k_sqrt_store(const float* acc, float* out) { *out = sqrtf(*acc); }

// ============================================================
// Host orchestration
// ============================================================
struct MabW {
  const _Float16 *WqT, *WkT, *WvT, *WoT, *WfT;   // pre-transposed f16 [N][K]
  const float *bq, *bk, *bv, *bo, *bf;
  const float *gx, *bx, *gy, *by, *gh, *bh;
};

extern "C" void kernel_launch(void* const* d_in, const int* in_sizes, int n_in,
                              void* d_out, int out_size, void* d_ws, size_t ws_size,
                              hipStream_t stream) {
  (void)in_sizes; (void)n_in; (void)out_size; (void)ws_size;
  const int B = MB, S = MS, F = MF_, D = MD, H = MH, L = ML;
  const int MEM = MMEM, NI = MNI, NCS = MNCS, NOUT = MOUT;

  // ---- inputs, flattened in setup_inputs() insertion order (recursively) ----
  const float* data       = (const float*)d_in[0];   // (B,S,F)
  const float* state      = (const float*)d_in[1];   // (L,B,MEM,D)
  const float* compressed = (const float*)d_in[2];   // (L,B,NCS,D)
  const float* enc_W = (const float*)d_in[3];
  const float* enc_b = (const float*)d_in[4];
  const float* lin_W = (const float*)d_in[5];
  const float* lin_b = (const float*)d_in[6];
  const float* isab_I = (const float*)d_in[7];       // (L,NI,D)
  const float* m1[16]; for (int i = 0; i < 16; i++) m1[i] = (const float*)d_in[8 + i];
  const float* m2[16]; for (int i = 0; i < 16; i++) m2[i] = (const float*)d_in[24 + i];
  const float* comp_S = (const float*)d_in[40];      // (L,NCS,D)
  const float* cm[16]; for (int i = 0; i < 16; i++) cm[i] = (const float*)d_in[41 + i];
  const float* pool_S = (const float*)d_in[57];      // (1,D)
  const float* pm[16]; for (int i = 0; i < 16; i++) pm[i] = (const float*)d_in[58 + i];
  const float* out_W = (const float*)d_in[74];       // (D,MOUT)
  const float* out_b = (const float*)d_in[75];
  float* outp = (float*)d_out;                        // probs (B*MOUT) then loss

  // ---- workspace layout ----
  float* ws = (float*)d_ws;
  size_t off = 0;
  auto alloc = [&](size_t elems) { float* p = ws + off; off += (elems + 255) & ~(size_t)255; return p; };
  auto alloc16 = [&](size_t halves) { return (_Float16*)alloc((halves + 1) / 2); };
  const size_t SZ = (size_t)B * S * D;   // 4.19M elements
  float*    contentA = alloc(SZ);
  float*    contentB = alloc(SZ);
  float*    excess   = alloc(SZ);
  float*    memsbuf  = alloc((size_t)B * (MEM + NCS) * D);
  float*    unc      = alloc(SZ);
  float*    compatt  = alloc(SZ);
  float*    compex   = alloc((size_t)B * NCS * D);
  float*    hb       = alloc((size_t)B * NI * D);
  float*    att      = alloc((size_t)B * H * 64 * 512 + 4096);
  float*    hbuf     = alloc(SZ);
  float*    pooled   = alloc((size_t)B * D);
  float*    logits   = alloc((size_t)B * NOUT);
  float*    lossacc  = alloc(16);
  _Float16* lnx16  = alloc16(SZ);
  _Float16* lny16  = alloc16(SZ);
  _Float16* qbuf16 = alloc16(SZ);   // also reused as encoder activation
  _Float16* kbuf16 = alloc16(SZ);
  _Float16* vT16   = alloc16(SZ);
  _Float16* obuf16 = alloc16(SZ);
  _Float16* lnh16  = alloc16(SZ);

  // ---- launch helpers ----
  auto gemm = [&](const void* A, int lda, long long sAo, long long sAi,
                  const void* Bmat, int ldb, long long sBo, long long sBi,
                  void* C, int ldc, long long sCo, long long sCi,
                  const float* bias, const float* res, int ldres,
                  long long sRo, long long sRi,
                  int M, int N, int K, int batches, int inner,
                  float alpha, int flags) {
    GemmP p;
    p.A = A; p.Bm = Bmat; p.bias = bias; p.res = res; p.C = C;
    p.M = M; p.N = N; p.K = K;
    p.lda = lda; p.ldb = ldb; p.ldc = ldc; p.ldres = ldres;
    p.sAo = sAo; p.sAi = sAi; p.sBo = sBo; p.sBi = sBi;
    p.sCo = sCo; p.sCi = sCi; p.sRo = sRo; p.sRi = sRi;
    p.inner = inner; p.alpha = alpha; p.flags = flags;
    dim3 grid((N + BN - 1) / BN, (M + BM - 1) / BM, batches);
    const bool a16 = (flags & FA16) != 0, b16 = (flags & FB16) != 0;
    if (a16 && b16)       k_gemm<true,  true ><<<grid, dim3(256), 0, stream>>>(p);
    else if (!a16 && b16) k_gemm<false, true ><<<grid, dim3(256), 0, stream>>>(p);
    else if (a16 && !b16) k_gemm<true,  false><<<grid, dim3(256), 0, stream>>>(p);
    else                  k_gemm<false, false><<<grid, dim3(256), 0, stream>>>(p);
  };
  auto tW = [&](const float* W, int K, int N) {   // -> f16 [N][K]
    _Float16* wt = alloc16((size_t)N * K);
    long long total = (long long)K * N;
    k_transpose16<<<(int)((total + 255) / 256), 256, 0, stream>>>(W, wt, K, N);
    return (const _Float16*)wt;
  };
  auto copy_rows = [&](float* dst, long long dstBS, int dstRowOff,
                       const float* src, long long srcBS, int Bn, int R) {
    long long total = (long long)Bn * R * D;
    k_copy_rows<<<(int)((total + 255) / 256), 256, 0, stream>>>(dst, dstBS, dstRowOff,
                                                                src, srcBS, Bn, R, D);
  };

  // ---- pre-transpose all weights to f16 [N][K] (once; L2-resident after) ----
  auto mabw = [&](const float* const* a, int layer) {
    size_t om = (size_t)layer * D * D, ov = (size_t)layer * D;
    MabW w;
    w.WqT = tW(a[0] + om, D, D);  w.bq = a[1] + ov;
    w.WkT = tW(a[2] + om, D, D);  w.bk = a[3] + ov;
    w.WvT = tW(a[4] + om, D, D);  w.bv = a[5] + ov;
    w.WoT = tW(a[6] + om, D, D);  w.bo = a[7] + ov;
    w.WfT = tW(a[8] + om, D, D);  w.bf = a[9] + ov;
    w.gx = a[10] + ov; w.bx = a[11] + ov;
    w.gy = a[12] + ov; w.by = a[13] + ov;
    w.gh = a[14] + ov; w.bh = a[15] + ov;
    return w;
  };
  MabW W1[ML], W2[ML], WC[ML];
  for (int i = 0; i < L; i++) { W1[i] = mabw(m1, i); W2[i] = mabw(m2, i); WC[i] = mabw(cm, i); }
  MabW WP = mabw(pm, 0);
  const _Float16* encT = tW(enc_W, F, D);       // [D][F]
  const _Float16* linT = tW(lin_W, D, D);       // [D][D]
  const _Float16* outT = tW(out_W, D, NOUT);    // [NOUT][D]

  // ---- full MAB block: out = h + relu(LN(h)@Wf+bf), h = x + MHA(LN(x),LN(y)) ----
  auto mab = [&](const float* x, int Nq, bool xShared,
                 const float* y, int Nk, const MabW& w, float* outbuf) {
    const int qRows = xShared ? Nq : B * Nq;
    const long long qbs = xShared ? 0 : (long long)Nq * D;
    k_layernorm16<<<qRows, 128, 0, stream>>>(x, w.gx, w.bx, lnx16, D);
    k_layernorm16<<<B * Nk, 128, 0, stream>>>(y, w.gy, w.by, lny16, D);
    // projections (weights shared across batch -> single large GEMM)
    gemm(lnx16, D, 0, 0, w.WqT, D, 0, 0, qbuf16, D, 0, 0, w.bq, nullptr, 0, 0, 0,
         qRows, D, D, 1, 1, 1.f, FA16 | FB16 | FC16 | FBIAS);
    gemm(lny16, D, 0, 0, w.WkT, D, 0, 0, kbuf16, D, 0, 0, w.bk, nullptr, 0, 0, 0,
         B * Nk, D, D, 1, 1, 1.f, FA16 | FB16 | FC16 | FBIAS);
    // v stored transposed per batch: vT[b, d, kk]  (so att@v has K-contiguous B)
    gemm(lny16, D, (long long)Nk * D, 0, w.WvT, D, 0, 0,
         vT16, Nk, (long long)D * Nk, 0, w.bv, nullptr, 0, 0, 0,
         Nk, D, D, B, 1, 1.f, FA16 | FB16 | FC16 | FTRANSC | FBIAS);
    // scores = (q @ k^T) / sqrt(dh), batched over (b,h)
    gemm(qbuf16, D, qbs, 64,
         kbuf16, D, (long long)Nk * D, 64,
         att, Nk, (long long)H * Nq * Nk, (long long)Nq * Nk,
         nullptr, nullptr, 0, 0, 0,
         Nq, Nk, MDH, B * H, H, 0.125f, FA16 | FB16);
    k_softmax<<<B * H * Nq, 256, 0, stream>>>(att, att, Nk);
    // o = att @ v, batched over (b,h), into head slices of (B,Nq,D) f16
    gemm(att, Nk, (long long)H * Nq * Nk, (long long)Nq * Nk,
         vT16, Nk, (long long)D * Nk, (long long)MDH * Nk,
         obuf16, D, (long long)Nq * D, 64,
         nullptr, nullptr, 0, 0, 0,
         Nq, MDH, Nk, B * H, H, 1.f, FB16 | FC16);
    // h = x + o @ Wo + bo  (batched over B so broadcast-x residual works)
    gemm(obuf16, D, (long long)Nq * D, 0, w.WoT, D, 0, 0,
         hbuf, D, (long long)Nq * D, 0, w.bo, x, D, qbs, 0,
         Nq, D, D, B, 1, 1.f, FA16 | FB16 | FBIAS | FRES);
    k_layernorm16<<<B * Nq, 128, 0, stream>>>(hbuf, w.gh, w.bh, lnh16, D);
    // out = h + relu(LN(h) @ Wf + bf)
    gemm(lnh16, D, 0, 0, w.WfT, D, 0, 0, outbuf, D, 0, 0,
         w.bf, hbuf, D, 0, 0,
         B * Nq, D, D, 1, 1, 1.f, FA16 | FB16 | FBIAS | FRELU | FRES);
  };

  auto isab = [&](const float* x, int Nx, const float* mems, int Nm,
                  const float* Iptr, const MabW& w1, const MabW& w2, float* outbuf) {
    mab(Iptr, NI, true, mems, Nm, w1, hb);      // h  (B,NI,D)
    mab(x, Nx, false, hb, NI, w2, outbuf);      // out(B,Nx,D)
  };

  // ---- encoder: x = relu(data@enc_W+enc_b); content = x@lin_W+lin_b ----
  gemm(data, F, 0, 0, encT, F, 0, 0, qbuf16, D, 0, 0, enc_b, nullptr, 0, 0, 0,
       B * S, D, F, 1, 1, 1.f, FB16 | FC16 | FBIAS | FRELU);
  gemm(qbuf16, D, 0, 0, linT, D, 0, 0, contentA, D, 0, 0, lin_b, nullptr, 0, 0, 0,
       B * S, D, D, 1, 1, 1.f, FA16 | FB16 | FBIAS);

  float* cur = contentA;
  float* nxt = contentB;
  for (int i = 0; i < L; i++) {
    const float* st = state + (long long)i * B * MEM * D;
    const float* cp = compressed + (long long)i * B * NCS * D;
    const float* Ii = isab_I + (long long)i * NI * D;
    const float* Si = comp_S + (long long)i * NCS * D;

    if (i == L - 1) {
      // loss path is dead code for i < L-1 (loss is reassigned each layer)
      // excess = concat(state[i], content[:, :MEM])  -> (B, 2*MEM, D)
      copy_rows(excess, (long long)2 * MEM * D, 0,   st,  (long long)MEM * D, B, MEM);
      copy_rows(excess, (long long)2 * MEM * D, MEM, cur, (long long)S * D,   B, MEM);
      isab(cur, S, excess, 2 * MEM, Ii, W1[i], W2[i], unc);
      mab(Si, NCS, true, excess, 2 * MEM, WC[i], compex);      // PMA(excess)
      isab(cur, S, compex, NCS, Ii, W1[i], W2[i], compatt);
      k_zero1<<<1, 1, 0, stream>>>(lossacc);
      k_sumsq_diff<<<1024, 256, 0, stream>>>(unc, compatt, lossacc, (long long)B * S * D);
      k_sqrt_store<<<1, 1, 0, stream>>>(lossacc, outp + B * NOUT);
    }

    // mems = concat(state[i], compressed[i]) -> (B, MEM+NCS, D)
    copy_rows(memsbuf, (long long)(MEM + NCS) * D, 0,   st, (long long)MEM * D, B, MEM);
    copy_rows(memsbuf, (long long)(MEM + NCS) * D, MEM, cp, (long long)NCS * D, B, NCS);
    isab(cur, S, memsbuf, MEM + NCS, Ii, W1[i], W2[i], nxt);
    float* tmp = cur; cur = nxt; nxt = tmp;
  }

  // ---- pooling + classifier ----
  mab(pool_S, 1, true, cur, S, WP, pooled);   // (B,1,D)
  gemm(pooled, D, 0, 0, outT, D, 0, 0, logits, NOUT, 0, 0,
       out_b, nullptr, 0, 0, 0,
       B, NOUT, D, 1, 1, 1.f, FB16 | FBIAS);
  k_softmax<<<B, 256, 0, stream>>>(logits, outp, NOUT);
}
